// Swarm_12979391168885
// MI455X (gfx1250) — compile-verified, matrix operand order checked by reading, not run on hardware
//
#include <hip/hip_runtime.h>

// ---------------------------------------------------------------------------
// CDNA5 (gfx1250) wave32 WMMA bf16 implementation of the swarm transformer.
// ---------------------------------------------------------------------------

typedef __attribute__((ext_vector_type(16))) __bf16 v16bf;
typedef __attribute__((ext_vector_type(8)))  float  v8f;

#define S_  8
#define N_  256
#define B_  64
#define H_  256
#define L_  64
#define NH_ 4
#define HD_ 64
#define MF_ 256
#define NBL (N_ * B_ * L_)

// Load a 16x32 bf16 WMMA fragment's per-lane data. `p` points at element
// (k0 + g*8) of this lane's row; elements 0..7 = K[g*8 .. g*8+7],
// elements 8..15 = K[16+g*8 .. 16+g*8+7]  (ISA 7.12.2 16-bit A layout).
__device__ __forceinline__ v16bf ld_frag(const __bf16* p) {
  v16bf r;
  ((uint4*)&r)[0] = *(const uint4*)(p);
  ((uint4*)&r)[1] = *(const uint4*)(p + 16);
  return r;
}

__device__ __forceinline__ v8f wmma_bf16(v16bf a, v16bf b, v8f c) {
  return __builtin_amdgcn_wmma_f32_16x16x32_bf16(false, a, false, b, (short)0, c,
                                                 false, false);
}

// One wave computes a (MT*16) x (NT*16) C tile of A(row-major bf16, lda) @ W^T,
// where W is row-major (out x K). K must be a multiple of 32.
template <int MT, int NT>
__device__ __forceinline__ void wave_gemm(const __bf16* A, int lda,
                                          const __bf16* W, int ldw, int K,
                                          int m0, int n0, int lane,
                                          v8f (&C)[MT][NT]) {
  const int mr = lane & 15, g = lane >> 4;
  for (int k0 = 0; k0 < K; k0 += 32) {
    v16bf a[MT], b[NT];
#pragma unroll
    for (int i = 0; i < MT; ++i)
      a[i] = ld_frag(A + (m0 + i * 16 + mr) * lda + k0 + g * 8);
#pragma unroll
    for (int j = 0; j < NT; ++j)
      b[j] = ld_frag(W + (n0 + j * 16 + mr) * ldw + k0 + g * 8);
#pragma unroll
    for (int i = 0; i < MT; ++i)
#pragma unroll
      for (int j = 0; j < NT; ++j) C[i][j] = wmma_bf16(a[i], b[j], C[i][j]);
  }
}

// ---------------------------------------------------------------------------
__global__ void k_cast_bf16(const float* __restrict__ src, __bf16* __restrict__ dst, int n) {
  int i = blockIdx.x * blockDim.x + threadIdx.x;
  if (i < n) dst[i] = (__bf16)src[i];
}

// qvec[s] = LN(bq[s]) * gq + betaq   (query == zeros -> species_linear == bias)
__global__ void k_qvec(const float* __restrict__ bq, const float* __restrict__ gq,
                       const float* __restrict__ betaq, float* __restrict__ qvec) {
  int s = blockIdx.x, t = threadIdx.x;
  __shared__ float red[256];
  float x = bq[s * 256 + t];
  red[t] = x; __syncthreads();
  for (int off = 128; off > 0; off >>= 1) { if (t < off) red[t] += red[t + off]; __syncthreads(); }
  float mu = red[0] * (1.f / 256.f);
  __syncthreads();
  float d = x - mu;
  red[t] = d * d; __syncthreads();
  for (int off = 128; off > 0; off >>= 1) { if (t < off) red[t] += red[t + off]; __syncthreads(); }
  float var = red[0] * (1.f / 256.f);
  qvec[s * 256 + t] = d * rsqrtf(var + 1e-5f) * gq[t] + betaq[t];
}

// qp_bf[s][o] = (qvec[s] . in_w[o,:] + in_b[o]) * 0.125   (1/sqrt(HD) folded)
__global__ void k_qp(const float* __restrict__ qvec, const float* __restrict__ in_w,
                     const float* __restrict__ in_b, __bf16* __restrict__ qp_bf) {
  int s = blockIdx.x, o = threadIdx.x;
  __shared__ float sq[256];
  sq[o] = qvec[s * 256 + o];
  __syncthreads();
  float acc = in_b[o];
  const float* w = in_w + o * 256;
  for (int i = 0; i < 256; ++i) acc += sq[i] * w[i];
  qp_bf[s * 256 + o] = (__bf16)(acc * 0.125f);
}

// Per-n: k = LN(msgs@wk^T+bk); kp = k@in_w_k^T+in_b_k -> k_bf (B,NH,N,HD)
//        v likewise -> vT_bf (B,NH,HD,N)
__global__ void k_kv(const float* __restrict__ all_msgs, const int* __restrict__ si,
                     const __bf16* __restrict__ wk_bf, const float* __restrict__ bk,
                     const float* __restrict__ gk, const float* __restrict__ betak,
                     const __bf16* __restrict__ inwk_bf,
                     const __bf16* __restrict__ wv_bf, const float* __restrict__ bv,
                     const float* __restrict__ gv, const float* __restrict__ betav,
                     const __bf16* __restrict__ inwv_bf,
                     const float* __restrict__ in_b,
                     __bf16* __restrict__ k_out, __bf16* __restrict__ vT_out) {
  extern __shared__ char smem[];
  __bf16* sA = (__bf16*)smem;                    // 64x256 bf16 (32KB)
  float*  sT = (float*)(smem + 32768);           // 64x256 f32  (64KB)
  __bf16* sL = (__bf16*)(smem + 32768 + 65536);  // 64x256 bf16 (32KB)
  __shared__ float redS[64][4], redQ[64][4];

  const int n = blockIdx.x;
  const int s = si[n];
  const int tid = threadIdx.x;
  const int w = tid >> 5, lane = tid & 31;
  const int gg = lane >> 4, nn = lane & 15;

  for (int i = tid; i < 64 * 256; i += 256)
    sA[i] = (__bf16)all_msgs[(size_t)n * 16384 + i];
  __syncthreads();

  for (int pass = 0; pass < 2; ++pass) {
    const __bf16* W1 = (pass ? wv_bf : wk_bf) + (size_t)s * 65536;
    const float* b1 = (pass ? bv : bk) + s * 256;
    const float* g  = pass ? gv : gk;
    const float* be = pass ? betav : betak;
    const __bf16* W2 = pass ? inwv_bf : inwk_bf;
    const float* b2 = in_b + (pass ? 512 : 256);

    {  // GEMM1: sA @ W1^T + b1 -> sT (f32)
      int m0 = (w >> 2) * 32, n0 = (w & 3) * 64;
      v8f C[2][4] = {};
      wave_gemm<2, 4>(sA, 256, W1, 256, 256, m0, n0, lane, C);
#pragma unroll
      for (int i = 0; i < 2; ++i)
#pragma unroll
        for (int j = 0; j < 4; ++j) {
          int col = n0 + j * 16 + nn;
          float bb = b1[col];
#pragma unroll
          for (int r = 0; r < 8; ++r)
            sT[(m0 + i * 16 + r + 8 * gg) * 256 + col] = C[i][j][r] + bb;
        }
    }
    __syncthreads();

    {  // row LN: sT -> sL (bf16)
      int row = tid >> 2, p = tid & 3;
      float sum = 0.f, sq = 0.f;
      const float* rp = sT + row * 256 + p * 64;
      for (int i = 0; i < 64; ++i) { float x = rp[i]; sum += x; sq += x * x; }
      redS[row][p] = sum; redQ[row][p] = sq;
      __syncthreads();
      float ts = redS[row][0] + redS[row][1] + redS[row][2] + redS[row][3];
      float tq = redQ[row][0] + redQ[row][1] + redQ[row][2] + redQ[row][3];
      float mu = ts * (1.f / 256.f);
      float var = tq * (1.f / 256.f) - mu * mu;
      float inv = rsqrtf(var + 1e-5f);
      __bf16* op = sL + row * 256 + p * 64;
      const float* gp = g + p * 64;
      const float* bp = be + p * 64;
      for (int i = 0; i < 64; ++i)
        op[i] = (__bf16)((rp[i] - mu) * inv * gp[i] + bp[i]);
    }
    __syncthreads();

    {  // GEMM2: sL @ W2^T + b2 -> scatter to k_bf / vT_bf layouts
      int m0 = (w >> 2) * 32, n0 = (w & 3) * 64;
      v8f C[2][4] = {};
      wave_gemm<2, 4>(sL, 256, W2, 256, 256, m0, n0, lane, C);
#pragma unroll
      for (int i = 0; i < 2; ++i)
#pragma unroll
        for (int j = 0; j < 4; ++j) {
          int col = n0 + j * 16 + nn;
          int head = col >> 6, d = col & 63;
          float bb = b2[col];
#pragma unroll
          for (int r = 0; r < 8; ++r) {
            int b = m0 + i * 16 + r + 8 * gg;
            float v = C[i][j][r] + bb;
            if (pass == 0)
              k_out[(((size_t)b * 4 + head) * 256 + n) * 64 + d] = (__bf16)v;
            else
              vT_out[(((size_t)b * 4 + head) * 64 + d) * 256 + n] = (__bf16)v;
          }
        }
    }
    __syncthreads();
  }
}

// Per (b, head, q-strip of 64): scores -> masked softmax -> ctx
__global__ void k_attn(const __bf16* __restrict__ qp_bf, const int* __restrict__ si,
                       const __bf16* __restrict__ k_bf, const __bf16* __restrict__ vT_bf,
                       const float* __restrict__ mask, __bf16* __restrict__ ctx_bf) {
  extern __shared__ char smem[];
  float*  sS  = (float*)smem;                 // 64x256 f32 (64KB)
  __bf16* sAW = (__bf16*)(smem + 65536);      // 64x256 bf16 (32KB)
  __shared__ float redM[64][4], redSum[64][4];

  const int blk = blockIdx.x;
  const int b = blk >> 4;
  const int head = (blk >> 2) & 3;
  const int q0 = (blk & 3) * 64;
  const int tid = threadIdx.x;
  const int w = tid >> 5, lane = tid & 31;
  const int mr = lane & 15, g = lane >> 4;
  const __bf16* Kb = k_bf + ((size_t)b * 4 + head) * 256 * 64;  // (kn, d)

  {  // scores (64 x 256), K = HD = 64
    int m0 = (w >> 2) * 32, n0 = (w & 3) * 64;
    v8f C[2][4] = {};
    for (int k0 = 0; k0 < 64; k0 += 32) {
      v16bf a[2], bf[4];
#pragma unroll
      for (int i = 0; i < 2; ++i) {
        int qn = q0 + m0 + i * 16 + mr;
        a[i] = ld_frag(qp_bf + si[qn] * 256 + head * 64 + k0 + g * 8);
      }
#pragma unroll
      for (int j = 0; j < 4; ++j) {
        int kn = n0 + j * 16 + mr;
        bf[j] = ld_frag(Kb + kn * 64 + k0 + g * 8);
      }
#pragma unroll
      for (int i = 0; i < 2; ++i)
#pragma unroll
        for (int j = 0; j < 4; ++j) C[i][j] = wmma_bf16(a[i], bf[j], C[i][j]);
    }
#pragma unroll
    for (int i = 0; i < 2; ++i)
#pragma unroll
      for (int j = 0; j < 4; ++j) {
        int col = n0 + j * 16 + mr;
#pragma unroll
        for (int r = 0; r < 8; ++r)
          sS[(m0 + i * 16 + r + 8 * g) * 256 + col] = C[i][j][r];
      }
  }
  __syncthreads();

  {  // masked softmax over kn, write bf16 attention weights
    int row = tid >> 2, p = tid & 3;
    int qn = q0 + row;
    const float* mrow = mask + (size_t)qn * 256 + p * 64;
    float* rp = sS + row * 256 + p * 64;
    float mx = -1e30f;
    for (int i = 0; i < 64; ++i) {
      float v = (mrow[i] == 1.0f) ? rp[i] : -1e9f;
      rp[i] = v;
      mx = fmaxf(mx, v);
    }
    redM[row][p] = mx; __syncthreads();
    float tmx = fmaxf(fmaxf(redM[row][0], redM[row][1]),
                      fmaxf(redM[row][2], redM[row][3]));
    float sum = 0.f;
    for (int i = 0; i < 64; ++i) { float e = __expf(rp[i] - tmx); rp[i] = e; sum += e; }
    redSum[row][p] = sum; __syncthreads();
    float inv = 1.0f / (redSum[row][0] + redSum[row][1] + redSum[row][2] + redSum[row][3]);
    __bf16* ap = sAW + row * 256 + p * 64;
    for (int i = 0; i < 64; ++i) ap[i] = (__bf16)(rp[i] * inv);
  }
  __syncthreads();

  {  // ctx = aw @ v : (64 x 64), K = 256
    int m0 = (w >> 1) * 16, n0 = (w & 1) * 32;
    v8f C[1][2] = {};
    const __bf16* Vt = vT_bf + ((size_t)b * 4 + head) * 64 * 256;  // (d, kn)
    wave_gemm<1, 2>(sAW, 256, Vt, 256, 256, m0, n0, lane, C);
#pragma unroll
    for (int j = 0; j < 2; ++j) {
      int col = n0 + j * 16 + mr;
#pragma unroll
      for (int r = 0; r < 8; ++r) {
        int qr = m0 + r + 8 * g;
        ctx_bf[((size_t)b * 256 + q0 + qr) * 256 + head * 64 + col] = (__bf16)C[0][j][r];
      }
    }
  }
}

// attn_out = ctx @ outw^T + outb  (per (b, 64-row strip))
__global__ void k_outproj(const __bf16* __restrict__ ctx_bf, const __bf16* __restrict__ outw_bf,
                          const float* __restrict__ outb, __bf16* __restrict__ attnout_bf) {
  extern __shared__ char smem[];
  __bf16* sA = (__bf16*)smem;  // 64x256 bf16
  const int blk = blockIdx.x;
  const int b = blk >> 2, r0 = (blk & 3) * 64;
  const int tid = threadIdx.x;
  const __bf16* src = ctx_bf + ((size_t)b * 256 + r0) * 256;
  for (int i = tid; i < 2048; i += 256) ((uint4*)sA)[i] = ((const uint4*)src)[i];
  __syncthreads();
  const int w = tid >> 5, lane = tid & 31;
  const int gg = lane >> 4, nn = lane & 15;
  int m0 = (w >> 2) * 32, n0 = (w & 3) * 64;
  v8f C[2][4] = {};
  wave_gemm<2, 4>(sA, 256, outw_bf, 256, 256, m0, n0, lane, C);
#pragma unroll
  for (int i = 0; i < 2; ++i)
#pragma unroll
    for (int j = 0; j < 4; ++j) {
      int col = n0 + j * 16 + nn;
      float bb = outb[col];
#pragma unroll
      for (int r = 0; r < 8; ++r) {
        int row = m0 + i * 16 + r + 8 * gg;
        attnout_bf[((size_t)b * 256 + r0 + row) * 256 + col] = (__bf16)(C[i][j][r] + bb);
      }
    }
}

// out = LN(attn_out(:,n,:) @ wo[s]^T + bo[s]) -> out_bf (n, b, h)
__global__ void k_out(const __bf16* __restrict__ attnout_bf, const int* __restrict__ si,
                      const __bf16* __restrict__ wo_bf, const float* __restrict__ bo,
                      const float* __restrict__ gout, const float* __restrict__ betaout,
                      __bf16* __restrict__ out_bf) {
  extern __shared__ char smem[];
  __bf16* sA = (__bf16*)smem;             // 64x256 bf16 (32KB)
  float*  sT = (float*)(smem + 32768);    // 64x256 f32  (64KB)
  __shared__ float redS[64][4], redQ[64][4];
  const int n = blockIdx.x, s = si[n], tid = threadIdx.x;
  for (int i = tid; i < 2048; i += 256) {
    int b = i >> 5, c = i & 31;
    ((uint4*)sA)[b * 32 + c] =
        ((const uint4*)(attnout_bf + ((size_t)b * 256 + n) * 256))[c];
  }
  __syncthreads();
  const int w = tid >> 5, lane = tid & 31;
  const int gg = lane >> 4, nn = lane & 15;
  {
    int m0 = (w >> 2) * 32, n0 = (w & 3) * 64;
    v8f C[2][4] = {};
    wave_gemm<2, 4>(sA, 256, wo_bf + (size_t)s * 65536, 256, 256, m0, n0, lane, C);
    const float* bb = bo + s * 256;
#pragma unroll
    for (int i = 0; i < 2; ++i)
#pragma unroll
      for (int j = 0; j < 4; ++j) {
        int col = n0 + j * 16 + nn;
        float bv = bb[col];
#pragma unroll
        for (int r = 0; r < 8; ++r)
          sT[(m0 + i * 16 + r + 8 * gg) * 256 + col] = C[i][j][r] + bv;
      }
  }
  __syncthreads();
  {
    int row = tid >> 2, p = tid & 3;
    float sum = 0.f, sq = 0.f;
    const float* rp = sT + row * 256 + p * 64;
    for (int i = 0; i < 64; ++i) { float x = rp[i]; sum += x; sq += x * x; }
    redS[row][p] = sum; redQ[row][p] = sq;
    __syncthreads();
    float ts = redS[row][0] + redS[row][1] + redS[row][2] + redS[row][3];
    float tq = redQ[row][0] + redQ[row][1] + redQ[row][2] + redQ[row][3];
    float mu = ts * (1.f / 256.f);
    float var = tq * (1.f / 256.f) - mu * mu;
    float inv = rsqrtf(var + 1e-5f);
    __bf16* op = out_bf + (size_t)n * 16384 + row * 256 + p * 64;
    const float* gp = gout + p * 64;
    const float* bp = betaout + p * 64;
    for (int i = 0; i < 64; ++i)
      op[i] = (__bf16)((rp[i] - mu) * inv * gp[i] + bp[i]);
  }
}

// FFN: hat_z = relu(out@w1^T+b1)@w2^T+b2 ; hat_z_next = relu(hat_z@w3^T+b3)@w4^T+b4
__global__ void k_ffn(const __bf16* __restrict__ out_bf, const int* __restrict__ si,
                      const __bf16* __restrict__ w1_bf, const float* __restrict__ b1,
                      const __bf16* __restrict__ w2_bf, const float* __restrict__ b2,
                      const __bf16* __restrict__ w3_bf, const float* __restrict__ b3,
                      const __bf16* __restrict__ w4_bf, const float* __restrict__ b4,
                      float* __restrict__ hatz_next, float* __restrict__ hatz) {
  extern __shared__ char smem[];
  __bf16* sA = (__bf16*)smem;              // 64x256
  __bf16* sH = (__bf16*)(smem + 32768);    // 64x256
  __bf16* sZ = (__bf16*)(smem + 65536);    // 64x64
  const int n = blockIdx.x, s = si[n], tid = threadIdx.x;
  for (int i = tid; i < 2048; i += 256)
    ((uint4*)sA)[i] = ((const uint4*)(out_bf + (size_t)n * 16384))[i];
  __syncthreads();
  const int w = tid >> 5, lane = tid & 31;
  const int gg = lane >> 4, nn = lane & 15;

  {  // h1 = relu(sA @ w1^T + b1) -> sH
    int m0 = (w >> 2) * 32, n0 = (w & 3) * 64;
    v8f C[2][4] = {};
    wave_gemm<2, 4>(sA, 256, w1_bf + (size_t)s * 65536, 256, 256, m0, n0, lane, C);
#pragma unroll
    for (int i = 0; i < 2; ++i)
#pragma unroll
      for (int j = 0; j < 4; ++j) {
        int col = n0 + j * 16 + nn;
        float bb = b1[s * 256 + col];
#pragma unroll
        for (int r = 0; r < 8; ++r)
          sH[(m0 + i * 16 + r + 8 * gg) * 256 + col] =
              (__bf16)fmaxf(C[i][j][r] + bb, 0.f);
      }
  }
  __syncthreads();

  {  // hat_z = sH @ w2^T + b2 -> global f32 + sZ bf16
    int m0 = (w >> 1) * 16, n0 = (w & 1) * 32;
    v8f C[1][2] = {};
    wave_gemm<1, 2>(sH, 256, w2_bf + (size_t)s * 16384, 256, 256, m0, n0, lane, C);
#pragma unroll
    for (int j = 0; j < 2; ++j) {
      int col = n0 + j * 16 + nn;
      float bb = b2[s * 64 + col];
#pragma unroll
      for (int r = 0; r < 8; ++r) {
        int row = m0 + r + 8 * gg;
        float v = C[0][j][r] + bb;
        hatz[((size_t)n * 64 + row) * 64 + col] = v;
        sZ[row * 64 + col] = (__bf16)v;
      }
    }
  }
  __syncthreads();

  {  // h2 = relu(sZ @ w3^T + b3) -> sH   (K = L = 64)
    int m0 = (w >> 2) * 32, n0 = (w & 3) * 64;
    v8f C[2][4] = {};
    wave_gemm<2, 4>(sZ, 64, w3_bf + (size_t)s * 16384, 64, 64, m0, n0, lane, C);
#pragma unroll
    for (int i = 0; i < 2; ++i)
#pragma unroll
      for (int j = 0; j < 4; ++j) {
        int col = n0 + j * 16 + nn;
        float bb = b3[s * 256 + col];
#pragma unroll
        for (int r = 0; r < 8; ++r)
          sH[(m0 + i * 16 + r + 8 * gg) * 256 + col] =
              (__bf16)fmaxf(C[i][j][r] + bb, 0.f);
      }
  }
  __syncthreads();

  {  // hat_z_next = sH @ w4^T + b4 -> global f32
    int m0 = (w >> 1) * 16, n0 = (w & 1) * 32;
    v8f C[1][2] = {};
    wave_gemm<1, 2>(sH, 256, w4_bf + (size_t)s * 16384, 256, 256, m0, n0, lane, C);
#pragma unroll
    for (int j = 0; j < 2; ++j) {
      int col = n0 + j * 16 + nn;
      float bb = b4[s * 64 + col];
#pragma unroll
      for (int r = 0; r < 8; ++r) {
        int row = m0 + r + 8 * gg;
        hatz_next[((size_t)n * 64 + row) * 64 + col] = C[0][j][r] + bb;
      }
    }
  }
}

// ---------------------------------------------------------------------------
extern "C" void kernel_launch(void* const* d_in, const int* in_sizes, int n_in,
                              void* d_out, int out_size, void* d_ws, size_t ws_size,
                              hipStream_t stream) {
  (void)in_sizes; (void)n_in; (void)out_size; (void)ws_size;
  const float* all_msgs = (const float*)d_in[0];
  const int*   si       = (const int*)d_in[1];
  const float* mask     = (const float*)d_in[2];
  const float* wk = (const float*)d_in[4];
  const float* wv = (const float*)d_in[5];
  const float* wo = (const float*)d_in[6];
  const float* w1 = (const float*)d_in[7];
  const float* w2 = (const float*)d_in[8];
  const float* w3 = (const float*)d_in[9];
  const float* w4 = (const float*)d_in[10];
  const float* bq = (const float*)d_in[11];
  const float* bk = (const float*)d_in[12];
  const float* bv = (const float*)d_in[13];
  const float* bo = (const float*)d_in[14];
  const float* b1 = (const float*)d_in[15];
  const float* b2 = (const float*)d_in[16];
  const float* b3 = (const float*)d_in[17];
  const float* b4 = (const float*)d_in[18];
  const float* gq = (const float*)d_in[19];
  const float* gk = (const float*)d_in[20];
  const float* gv = (const float*)d_in[21];
  const float* gout = (const float*)d_in[22];
  const float* betaq = (const float*)d_in[23];
  const float* betak = (const float*)d_in[24];
  const float* betav = (const float*)d_in[25];
  const float* betaout = (const float*)d_in[26];
  const float* in_w = (const float*)d_in[27];
  const float* in_b = (const float*)d_in[28];
  const float* outw = (const float*)d_in[29];
  const float* outb = (const float*)d_in[30];

  char* ws = (char*)d_ws;
  // byte offsets into workspace (all 256-aligned)
  const size_t WKB = 0;             // 8*256*256 bf16 = 1 MiB
  const size_t WVB = 1048576;
  const size_t WOB = 2097152;
  const size_t W1B = 3145728;
  const size_t W2B = 4194304;       // 8*64*256 bf16 = 256 KiB
  const size_t W3B = 4456448;
  const size_t W4B = 4718592;
  const size_t INWK = 4980736;      // 256*256 bf16 = 128 KiB
  const size_t INWV = 5111808;
  const size_t OUTW = 5242880;
  const size_t QVEC = 5373952;      // 8*256 f32
  const size_t QPB  = 5382144;      // 8*256 bf16
  const size_t KBF  = 5386240;      // B*NH*N*HD bf16 = 8 MiB
  const size_t VTB  = 13774848;
  const size_t CTX  = 22163456;
  const size_t AOUT = 30552064;
  const size_t OUTB = 38940672;

  __bf16* wk_bf   = (__bf16*)(ws + WKB);
  __bf16* wv_bf   = (__bf16*)(ws + WVB);
  __bf16* wo_bf   = (__bf16*)(ws + WOB);
  __bf16* w1_bf   = (__bf16*)(ws + W1B);
  __bf16* w2_bf   = (__bf16*)(ws + W2B);
  __bf16* w3_bf   = (__bf16*)(ws + W3B);
  __bf16* w4_bf   = (__bf16*)(ws + W4B);
  __bf16* inwk_bf = (__bf16*)(ws + INWK);
  __bf16* inwv_bf = (__bf16*)(ws + INWV);
  __bf16* outw_bf = (__bf16*)(ws + OUTW);
  float*  qvec    = (float*)(ws + QVEC);
  __bf16* qp_bf   = (__bf16*)(ws + QPB);
  __bf16* k_bf    = (__bf16*)(ws + KBF);
  __bf16* vT_bf   = (__bf16*)(ws + VTB);
  __bf16* ctx_bf  = (__bf16*)(ws + CTX);
  __bf16* aout_bf = (__bf16*)(ws + AOUT);
  __bf16* out_bf  = (__bf16*)(ws + OUTB);

  auto cast = [&](const float* src, __bf16* dst, int n) {
    k_cast_bf16<<<(n + 255) / 256, 256, 0, stream>>>(src, dst, n);
  };
  cast(wk, wk_bf, S_ * H_ * MF_);
  cast(wv, wv_bf, S_ * H_ * MF_);
  cast(wo, wo_bf, S_ * H_ * H_);
  cast(w1, w1_bf, S_ * H_ * H_);
  cast(w2, w2_bf, S_ * L_ * H_);
  cast(w3, w3_bf, S_ * H_ * L_);
  cast(w4, w4_bf, S_ * L_ * H_);
  cast(in_w + H_ * H_,     inwk_bf, H_ * H_);
  cast(in_w + 2 * H_ * H_, inwv_bf, H_ * H_);
  cast(outw, outw_bf, H_ * H_);

  k_qvec<<<S_, 256, 0, stream>>>(bq, gq, betaq, qvec);
  k_qp<<<S_, 256, 0, stream>>>(qvec, in_w, in_b, qp_bf);

  k_kv<<<N_, 256, 131072, stream>>>(all_msgs, si, wk_bf, bk, gk, betak, inwk_bf,
                                    wv_bf, bv, gv, betav, inwv_bf, in_b,
                                    k_bf, vT_bf);

  k_attn<<<B_ * NH_ * 4, 256, 98304, stream>>>(qp_bf, si, k_bf, vT_bf, mask, ctx_bf);

  k_outproj<<<B_ * 4, 256, 32768, stream>>>(ctx_bf, outw_bf, outb, aout_bf);

  k_out<<<N_, 256, 98304, stream>>>(aout_bf, si, wo_bf, bo, gout, betaout, out_bf);

  float* hatz_next = (float*)d_out;        // first tuple element
  float* hatz      = (float*)d_out + NBL;  // second tuple element
  k_ffn<<<N_, 256, 73728, stream>>>(out_bf, si, w1_bf, b1, w2_bf, b2,
                                    w3_bf, b3, w4_bf, b4, hatz_next, hatz);
}
